// QRNN3DLayer_14946486190553
// MI455X (gfx1250) — matrix operand order repaired and checked by reading, not compiled, not run on hardware
//
#include <hip/hip_runtime.h>
#include <hip/hip_bf16.h>

// ---------------------------------------------------------------------------
// QRNN3D fused kernel for MI455X (gfx1250), wave32 + WMMA bf16.
//
// conv3d(16->48, k=3, SAME) as implicit GEMM:
//   M = 16 spatial points (one h-row of a 16x16 HxW tile, M index = w)
//   N = 48 gate channels (3 blocks of 16: Z, F, O)
//   K = 27 taps * 16 in-channels = 432, padded to 448 (14 WMMA K-steps of 32)
// All 42 B-fragments (weights) live in VGPRs for the whole kernel; the
// fo-pool recurrence state c stays in the WMMA accumulator layout, so gates
// are never materialized in HBM (min traffic: 130MB in + 130MB out).
// ---------------------------------------------------------------------------

typedef __attribute__((ext_vector_type(16))) __bf16 v16bf;
typedef __attribute__((ext_vector_type(8)))  float  v8f;
typedef __attribute__((ext_vector_type(4)))  unsigned int u32x4;

union ABFrag { u32x4 q[2]; v16bf v; };

__device__ __forceinline__ unsigned short f2bf(float f) {
  unsigned int u = __float_as_uint(f);
  u += 0x7FFFu + ((u >> 16) & 1u);        // round-to-nearest-even
  return (unsigned short)(u >> 16);
}
__device__ __forceinline__ unsigned int pk2(float a, float b) {
  return (unsigned int)f2bf(a) | ((unsigned int)f2bf(b) << 16);
}

#define CIN   16
#define TT    31
#define HH    128
#define WWID  128
#define HW    (HH * WWID)
#define STRC  (TT * HW)          /* channel stride in input/output: 507904 */
#define KTOT  448                /* 28 taps * 16 ci; tap 27 is a zero pad */
#define SLICE (324 * 16)         /* one 18x18x16 halo slice, ushort elems */

// --- Kernel 1: pack conv weights into bf16 implicit-GEMM B matrix [K][48] ---
__global__ void prepack_weights(const float* __restrict__ W,
                                unsigned short* __restrict__ wpk) {
  int i = blockIdx.x * blockDim.x + threadIdx.x;
  if (i >= KTOT * 48) return;
  int k  = i / 48;
  int oc = i - k * 48;
  unsigned short v = 0;
  if (k < 432) {
    int tap = k >> 4, ci = k & 15;
    int dt = tap / 9, r9 = tap - dt * 9, dh = r9 / 3, dw = r9 - dh * 3;
    // W layout: (O=48, I=16, D=3, H=3, W=3)
    v = f2bf(W[(((oc * CIN + ci) * 3 + dt) * 3 + dh) * 3 + dw]);
  }
  { int kk = k & 31, e = (kk & 7) | ((kk >> 4) << 3), ln = (((kk >> 3) & 1) << 4) + (oc & 15);
    wpk[(k >> 5) * 1536 + ln * 48 + (oc >> 4) * 16 + e] = v; }   // wpk[step][lane=16h+n][nb][e], k table order
}

// --- Kernel 2: fused conv + gates + fo-pool recurrence --------------------
__launch_bounds__(256)
__global__ void qrnn3d_fused(const float* __restrict__ in,
                             const unsigned short* __restrict__ wpk,
                             const float* __restrict__ bias,
                             float* __restrict__ out) {
  // 3-slot ring of input time-slices, halo'd 18x18, channel-innermost bf16.
  __shared__ __align__(16) unsigned short sInF[3 * SLICE];     // 31104 B
  // output staging; row stride 257 dwords => conflict-free ch-strided writes
  __shared__ float sOut[16 * 257];                             // 16448 B

  const int tid  = threadIdx.x;
  const int wave = tid >> 5;
  const int lane = tid & 31;

  const int bx  = blockIdx.x;          // 4 * 8 * 8 = 256 blocks
  const int n   = bx >> 6;
  const int rem = bx & 63;
  const int h0  = (rem >> 3) << 4;
  const int w0  = (rem & 7) << 4;

  // D-fragment: N (= out channel within 16-block) = lane & 15
  const int   chl = lane & 15;
  const float bz  = bias[chl];
  const float bfg = bias[16 + chl];
  const float bo  = bias[32 + chl];

  // A-fragment lane decode (16-bit A 16x32 layout):
  //   lanes 0-15 : M = lane,    V0-3 = tap_a ci 0..7  (K0..7),  V4-7 = tap_b ci 0..7  (K16..23)
  //   lanes 16-31: M = lane-16, V0-3 = tap_a ci 8..15 (K8..15), V4-7 = tap_b ci 8..15 (K24..31)
  const int wwa = lane & 15;           // M = w within tile
  const int cba = (lane >> 4) * 8;     // ci base
  const int aoff0 = wave * 288 + wwa * 16 + cba;          // mtile 0 (hh = wave)
  const int aoff1 = aoff0 + 8 * 288;                      // mtile 1 (hh = wave+8)

  // --- preload ALL B fragments into VGPRs (loop-invariant, 42 x 32B) ------
  const unsigned short* wLane = wpk + lane * 48;          // B lane = 16h + n (column n, k half h)
  ABFrag Bfr[14][3];
  #pragma unroll
  for (int s = 0; s < 14; ++s)
    #pragma unroll
    for (int nb = 0; nb < 3; ++nb) {
      const u32x4* bp = (const u32x4*)(wLane + s * 32 * 48 + nb * 16);
      Bfr[s][nb].q[0] = bp[0];
      Bfr[s][nb].q[1] = bp[1];
    }

  // --- one-time halo position decode (<=2 positions per thread) -----------
  const float* inN = in + (size_t)n * CIN * STRC;
  const int xh0 = tid / 18, xw0 = tid - xh0 * 18;
  const int gh0 = h0 + xh0 - 1, gw0 = w0 + xw0 - 1;
  const bool ok0 = (gh0 >= 0) & (gh0 < HH) & (gw0 >= 0) & (gw0 < WWID);
  const int gof0 = gh0 * WWID + gw0;
  const int lof0 = tid * 16;
  const int p1   = tid + 256;                  // second position (tid < 68)
  const int xh1 = p1 / 18, xw1 = p1 - xh1 * 18;
  const int gh1 = h0 + xh1 - 1, gw1 = w0 + xw1 - 1;
  const bool ok1 = (gh1 >= 0) & (gh1 < HH) & (gw1 >= 0) & (gw1 < WWID);
  const int gof1 = gh1 * WWID + gw1;
  const int lof1 = p1 * 16;

  auto loadSlice = [&](int sb, int ts, bool tsv) {
    const float* gpt = inN + (size_t)(tsv ? ts : 0) * HW;
    {
      float a0 = 0.f, a1 = 0.f, a2 = 0.f, a3 = 0.f;
      #pragma unroll
      for (int cq = 0; cq < 4; ++cq) {
        if (ok0 & tsv) {
          const float* g = gpt + gof0;
          a0 = g[(size_t)(cq * 4 + 0) * STRC];
          a1 = g[(size_t)(cq * 4 + 1) * STRC];
          a2 = g[(size_t)(cq * 4 + 2) * STRC];
          a3 = g[(size_t)(cq * 4 + 3) * STRC];
        } else { a0 = a1 = a2 = a3 = 0.f; }
        uint2 p; p.x = pk2(a0, a1); p.y = pk2(a2, a3);
        *(uint2*)&sInF[sb + lof0 + cq * 4] = p;
      }
    }
    if (tid < 68) {
      float a0, a1, a2, a3;
      #pragma unroll
      for (int cq = 0; cq < 4; ++cq) {
        if (ok1 & tsv) {
          const float* g = gpt + gof1;
          a0 = g[(size_t)(cq * 4 + 0) * STRC];
          a1 = g[(size_t)(cq * 4 + 1) * STRC];
          a2 = g[(size_t)(cq * 4 + 2) * STRC];
          a3 = g[(size_t)(cq * 4 + 3) * STRC];
        } else { a0 = a1 = a2 = a3 = 0.f; }
        uint2 p; p.x = pk2(a0, a1); p.y = pk2(a2, a3);
        *(uint2*)&sInF[sb + lof1 + cq * 4] = p;
      }
    }
  };

  // rotating ring-slot bases (ushort-element offsets): slice t-1, t, t+1
  int bm1 = 2 * SLICE;   // slice -1 (zeros) preloaded into slot 2
  int b0c = 0;           // slice 0 into slot 0
  int bp1 = SLICE;       // slot 1 receives slice 1 at t=0

  loadSlice(bm1, 0, false);    // zero-fill
  loadSlice(b0c, 0, true);

  // precomputed staging addresses: element (reg e, lane) -> M = e+8*(lane>=16)
  float* sOutP = &sOut[chl * 257 + wave * 16 + ((lane >> 4) << 3)];
  // per-thread output pointer (writeback phase): (hh2,ww2) from tid
  float* outT = out + (size_t)n * CIN * STRC
                    + (size_t)(h0 + (tid >> 4)) * WWID + (w0 + (tid & 15));

  // recurrence state c, kept in C/D accumulator layout; two M-tiles per wave
  v8f cst[2];
  #pragma unroll
  for (int e = 0; e < 8; ++e) { cst[0][e] = 0.f; cst[1][e] = 0.f; }

  #pragma unroll 1
  for (int t = 0; t < TT; ++t) {
    loadSlice(bp1, t + 1, (t + 1) < TT);
    __syncthreads();

    // accumulators pre-seeded with the gate biases (uniform per lane)
    v8f acc[2][3];
    #pragma unroll
    for (int m = 0; m < 2; ++m)
      #pragma unroll
      for (int e = 0; e < 8; ++e) {
        acc[m][0][e] = bz; acc[m][1][e] = bfg; acc[m][2][e] = bo;
      }

    #pragma unroll
    for (int s = 0; s < 14; ++s) {
      const int ta  = 2 * s;
      const int tb  = (2 * s + 1 < 27) ? (2 * s + 1) : 26;  // dup tap; B rows zero
      const int dta = ta / 9, ra = ta % 9, dha = ra / 3, dwa = ra % 3;
      const int dtb = tb / 9, rb = tb % 9, dhb = rb / 3, dwb = rb % 3;
      const int sba = (dta == 0) ? bm1 : (dta == 1) ? b0c : bp1;
      const int sbb = (dtb == 0) ? bm1 : (dtb == 1) ? b0c : bp1;
      const int ca  = (dha * 18 + dwa) * 16;
      const int cb  = (dhb * 18 + dwb) * 16;

      ABFrag A0, A1;
      A0.q[0] = *(const u32x4*)&sInF[sba + aoff0 + ca];
      A0.q[1] = *(const u32x4*)&sInF[sbb + aoff0 + cb];
      A1.q[0] = *(const u32x4*)&sInF[sba + aoff1 + ca];
      A1.q[1] = *(const u32x4*)&sInF[sbb + aoff1 + cb];

      #pragma unroll
      for (int nb = 0; nb < 3; ++nb) {
        acc[0][nb] = __builtin_amdgcn_wmma_f32_16x16x32_bf16(
            false, A0.v, false, Bfr[s][nb].v, (short)0, acc[0][nb], false, false);
        acc[1][nb] = __builtin_amdgcn_wmma_f32_16x16x32_bf16(
            false, A1.v, false, Bfr[s][nb].v, (short)0, acc[1][nb], false, false);
      }
    }

    // gates + fo-pool step
    #pragma unroll
    for (int mtI = 0; mtI < 2; ++mtI) {
      v8f c = cst[mtI];
      #pragma unroll
      for (int e = 0; e < 8; ++e) {
        float zv = 2.f * __builtin_amdgcn_rcpf(1.f + __expf(-2.f * acc[mtI][0][e])) - 1.f;
        float fv = __builtin_amdgcn_rcpf(1.f + __expf(-acc[mtI][1][e]));
        float ov = __builtin_amdgcn_rcpf(1.f + __expf(-acc[mtI][2][e]));
        float cn = zv + fv * (c[e] - zv);           // f*c + (1-f)*z
        c[e] = cn;
        sOutP[mtI * 128 + e] = ov * cn;             // conflict-free staging
      }
      cst[mtI] = c;
    }
    __syncthreads();

    // coalesced writeback: 256 threads cover one channel's 16x16 tile per pass
    #pragma unroll
    for (int ch = 0; ch < CIN; ++ch)
      outT[(size_t)ch * STRC] = sOut[ch * 257 + tid];

    // rotate ring slots; advance output time pointer
    { int tmp = bm1; bm1 = b0c; b0c = bp1; bp1 = tmp; }
    outT += HW;
  }
}

extern "C" void kernel_launch(void* const* d_in, const int* in_sizes, int n_in,
                              void* d_out, int out_size, void* d_ws, size_t ws_size,
                              hipStream_t stream) {
  const float* in = (const float*)d_in[0];
  const float* W  = (const float*)d_in[1];
  const float* b  = (const float*)d_in[2];
  float* out = (float*)d_out;
  unsigned short* wpk = (unsigned short*)d_ws;   // 448*48*2 = 43,008 bytes

  prepack_weights<<<(KTOT * 48 + 255) / 256, 256, 0, stream>>>(W, wpk);
  qrnn3d_fused<<<4 * 8 * 8, 256, 0, stream>>>(in, wpk, b, out);
}
